// SelfAttention_16166256902602
// MI455X (gfx1250) — matrix operand order repaired and checked by reading, not compiled, not run on hardware
//
#include <hip/hip_runtime.h>

// ---------------------------------------------------------------------------
// CDNA5 (gfx1250) bidirectional attention-GRU forward.
// S=384 seq, B=64 batch, I=H=256. wave32, WMMA bf16 16x16x32, f32 accum.
// GEMMs: async-copy (ASYNCcnt) double-buffered LDS staging -> ds_load_b128
// fragments -> v_wmma_f32_16x16x32_bf16.
// ---------------------------------------------------------------------------

#define S_ 384
#define B_ 64
#define I_ 256
#define H_ 256
#define SB_ (S_ * B_)
#define PADW 40   // 32 bf16 + 8 pad -> 80B rows, conflict-free ds_load_b128

typedef __attribute__((ext_vector_type(16))) __bf16 v16bf;
typedef __attribute__((ext_vector_type(8)))  float  v8f;
typedef __attribute__((ext_vector_type(4)))  unsigned int u32x4;

union FragU { v16bf v; u32x4 q[2]; };

__device__ __forceinline__ unsigned short f2bf(float x) {
    unsigned int u = __float_as_uint(x);
    u += 0x7fffu + ((u >> 16) & 1u);            // round-to-nearest-even
    return (unsigned short)(u >> 16);
}
__device__ __forceinline__ float sigm(float x) { return 1.f / (1.f + __expf(-x)); }
__device__ __forceinline__ float tanh_f(float x) {
    x = fminf(15.f, fmaxf(-15.f, x));
    float e = __expf(2.f * x);
    return (e - 1.f) / (e + 1.f);
}

// B fragment straight from global W stored [N x K] row-major (i.e. B^T):
// lane n = lane&15, 16 contiguous K per lane, lane-half selects K block of 16.
__device__ __forceinline__ v16bf load_b_frag(const unsigned short* base, int ldw) {
    int lane = threadIdx.x & 31, n = lane & 15, kb = (lane >> 4) * 16;
    const unsigned short* p = base + (size_t)n * ldw + kb;
    FragU f;
    f.q[0] = *(const u32x4*)p;
    f.q[1] = *(const u32x4*)(p + 8);
    return f.v;
}
// A fragment from an LDS-staged [rows][PADW] k-slice (local k in [0,32)).
// lane<16: e0..15 -> K=0..15 ; lane>=16: K=16..31 (matches B fragments).
__device__ __forceinline__ v16bf lds_a_frag(const unsigned short (*t)[PADW], int row0) {
    int lane = threadIdx.x & 31, m = lane & 15, hi = lane >> 4;
    const unsigned short* p = &t[row0 + m][hi * 16];
    FragU f;
    f.q[0] = *(const u32x4*)p;
    f.q[1] = *(const u32x4*)(p + 8);
    return f.v;
}
// B fragment from an LDS-staged [rows][PADW] k-slice.
__device__ __forceinline__ v16bf lds_b_frag(const unsigned short (*t)[PADW], int row0) {
    int lane = threadIdx.x & 31, n = lane & 15, kb = (lane >> 4) * 16;
    const unsigned short* p = &t[row0 + n][kb];
    FragU f;
    f.q[0] = *(const u32x4*)p;
    f.q[1] = *(const u32x4*)(p + 8);
    return f.v;
}
__device__ __forceinline__ v8f wmma_bf16(v16bf a, v16bf b, v8f c) {
    return __builtin_amdgcn_wmma_f32_16x16x32_bf16(false, a, false, b, (short)0, c, false, false);
}
__device__ __forceinline__ void async_ld_b128(unsigned lds_addr, const void* gaddr) {
    asm volatile("global_load_async_to_lds_b128 %0, %1, off"
                 :: "v"(lds_addr), "v"((unsigned long long)(size_t)gaddr) : "memory");
}
__device__ __forceinline__ void wait_async0() {
    asm volatile("s_wait_asynccnt 0" ::: "memory");
}

// ---------------------------------------------------------------------------
// Tiled GEMM: C[M,N] = A[M,K](bf16 row-major) * W[N,K]^T(bf16) + bias.
// Block 256 thr = 8 waves; block tile 128x128, wave tile 32x64 (8 wmma tiles).
// A and W k-slices (128x32 bf16) async-copied to LDS, double-buffered.
// ---------------------------------------------------------------------------
__global__ __launch_bounds__(256) void gemm_bf16_nt(
    const unsigned short* __restrict__ A, const unsigned short* __restrict__ W,
    float* __restrict__ C, const float* __restrict__ bias,
    int K, int lda, int ldw, int ldc,
    long long bsA, long long bsW, long long bsC)
{
    __shared__ unsigned short at[2][128][PADW];
    __shared__ unsigned short wt[2][128][PADW];

    const int bz = blockIdx.z;
    A += bz * bsA; W += bz * bsW; C += bz * bsC;

    const int tid  = threadIdx.x;
    const int wave = tid >> 5, lane = tid & 31;
    const int wm = wave & 3, wn = wave >> 2;         // 4 waves along M, 2 along N
    const size_t blockM = (size_t)blockIdx.y * 128;
    const size_t blockN = (size_t)blockIdx.x * 128;

    // Stage one 128x32 k-slice of A and W into LDS buffer `buf` (async).
    auto stage = [&](int buf, int k0) {
        #pragma unroll
        for (int ss = 0; ss < 2; ++ss) {
            int s = tid + ss * 256;                  // 512 16B segments each
            int r = s >> 2, seg = s & 3;
            const unsigned short* ga = A + (blockM + r) * (size_t)lda + k0 + seg * 8;
            const unsigned short* gw = W + (blockN + r) * (size_t)ldw + k0 + seg * 8;
            async_ld_b128((unsigned)(size_t)&at[buf][r][seg * 8], ga);
            async_ld_b128((unsigned)(size_t)&wt[buf][r][seg * 8], gw);
        }
    };

    stage(0, 0);

    v8f acc[2][4];
    for (int mt = 0; mt < 2; ++mt)
        for (int j = 0; j < 4; ++j)
            for (int e = 0; e < 8; ++e) acc[mt][j][e] = 0.f;

    wait_async0();
    __syncthreads();

    for (int k0 = 0; k0 < K; k0 += 32) {
        const int cur = (k0 >> 5) & 1;
        if (k0 + 32 < K) stage(cur ^ 1, k0 + 32);    // overlap next slice with WMMAs

        v16bf a0 = lds_a_frag(at[cur], wm * 32);
        v16bf a1 = lds_a_frag(at[cur], wm * 32 + 16);
        #pragma unroll
        for (int j = 0; j < 4; ++j) {
            v16bf b = lds_b_frag(wt[cur], wn * 64 + 16 * j);
            acc[0][j] = wmma_bf16(a0, b, acc[0][j]);
            acc[1][j] = wmma_bf16(a1, b, acc[1][j]);
        }
        wait_async0();
        __syncthreads();
    }

    const int nn = lane & 15, hi = lane >> 4;
    #pragma unroll
    for (int j = 0; j < 4; ++j) {
        size_t col = blockN + wn * 64 + 16 * j + nn;
        float bv = bias ? bias[col] : 0.f;
        #pragma unroll
        for (int mt = 0; mt < 2; ++mt) {
            #pragma unroll
            for (int v = 0; v < 8; ++v) {
                size_t row = blockM + wm * 32 + mt * 16 + v + 8 * hi;
                C[row * (size_t)ldc + col] = acc[mt][j][v] + bv;
            }
        }
    }
}

// ---------------------------------------------------------------------------
// Small utility kernels
// ---------------------------------------------------------------------------
__global__ void lengths_kernel(const float* __restrict__ mask, int* __restrict__ len) {
    int b = threadIdx.x;
    if (b < B_) {
        int c = 0;
        for (int s = 0; s < S_; ++s) c += (mask[b * S_ + s] > 0.f) ? 1 : 0;
        len[b] = c;
    }
}

__global__ void cvt_bf16_kernel(const float* __restrict__ src, unsigned short* __restrict__ dst, int n) {
    int i = blockIdx.x * 256 + threadIdx.x;
    if (i < n) dst[i] = f2bf(src[i]);
}

// bf16-convert qt for this direction (flip-gathered for left), also build
// qt^T [b][i][s] for the h_alpha GEMM B-operand, and fill z's "pp" half (f32).
__global__ void cvt_qt_kernel(const float* __restrict__ qt, const int* __restrict__ len, int flip,
                              unsigned short* __restrict__ qtb, unsigned short* __restrict__ qtT,
                              float* __restrict__ z)
{
    int idx = blockIdx.x, t = idx >> 6, b = idx & 63, i = threadIdx.x;
    int src = t;
    if (flip) { int lb = len[b]; src = (t < lb) ? (lb - 1 - t) : t; }
    float v = qt[((size_t)src * B_ + b) * I_ + i];
    unsigned short bv = f2bf(v);
    qtb[((size_t)t * B_ + b) * I_ + i] = bv;
    qtT[((size_t)b * I_ + i) * S_ + t] = bv;
    z[((size_t)t * B_ + b) * (2 * I_) + i] = v;
}

// score[t,b,s] = bv + sum_h Wv[h]*tanh(wp[s,b,h] + wpp[t,b,h])
__global__ __launch_bounds__(256) void score_kernel(
    const float* __restrict__ wp, const float* __restrict__ wpp,
    const float* __restrict__ Wv, const float* __restrict__ bv,
    float* __restrict__ scoref)
{
    int b = blockIdx.x, t = blockIdx.y;
    __shared__ float wppS[H_], wvS[H_];
    int tid = threadIdx.x;
    wppS[tid] = wpp[((size_t)t * B_ + b) * H_ + tid];
    wvS[tid]  = Wv[tid];
    __syncthreads();
    int lane = tid & 31, wave = tid >> 5;
    float bvv = bv[0];
    for (int s = wave; s < S_; s += 8) {
        const float* wr = wp + ((size_t)s * B_ + b) * H_ + lane * 8;
        float acc = 0.f;
        #pragma unroll
        for (int j = 0; j < 8; ++j)
            acc += tanh_f(wr[j] + wppS[lane * 8 + j]) * wvS[lane * 8 + j];
        #pragma unroll
        for (int off = 16; off; off >>= 1) acc += __shfl_xor(acc, off, 32);
        if (lane == 0) scoref[((size_t)t * B_ + b) * S_ + s] = acc + bvv;
    }
}

// masked softmax over s per (t,b) row; output bf16 alpha. One wave per row.
__global__ __launch_bounds__(256) void softmax_kernel(
    const float* __restrict__ scoref, const float* __restrict__ mask,
    unsigned short* __restrict__ alpha)
{
    int r = blockIdx.x * 8 + (threadIdx.x >> 5);       // r = t*64 + b
    int lane = threadIdx.x & 31, b = r & 63;
    const float* row = scoref + (size_t)r * S_;
    float xv[12], mx = -3.4e38f;
    #pragma unroll
    for (int j = 0; j < 12; ++j) {
        int s = j * 32 + lane;
        float x = row[s] + ((mask[(size_t)b * S_ + s] > 0.f) ? 0.f : -1e9f);
        xv[j] = x; mx = fmaxf(mx, x);
    }
    #pragma unroll
    for (int off = 16; off; off >>= 1) mx = fmaxf(mx, __shfl_xor(mx, off, 32));
    float sum = 0.f;
    #pragma unroll
    for (int j = 0; j < 12; ++j) { xv[j] = __expf(xv[j] - mx); sum += xv[j]; }
    #pragma unroll
    for (int off = 16; off; off >>= 1) sum += __shfl_xor(sum, off, 32);
    float inv = __fdividef(1.f, sum);
    #pragma unroll
    for (int j = 0; j < 12; ++j)
        alpha[(size_t)r * S_ + j * 32 + lane] = f2bf(xv[j] * inv);
}

// z = sigmoid(gl)*z ; layernorm(gamma,beta) ; store bf16 into zb. Wave/row.
__global__ __launch_bounds__(256) void gate_ln_kernel(
    const float* __restrict__ z, const float* __restrict__ gl,
    const float* __restrict__ gamma, const float* __restrict__ beta,
    unsigned short* __restrict__ zb)
{
    int r = blockIdx.x * 8 + (threadIdx.x >> 5);
    int lane = threadIdx.x & 31;
    const float* zr = z + (size_t)r * 512;
    const float* gr = gl + (size_t)r * 512;
    float v[16], sum = 0.f;
    #pragma unroll
    for (int j = 0; j < 16; ++j) {
        int c = j * 32 + lane;
        float zi = zr[c] * sigm(gr[c]);
        v[j] = zi; sum += zi;
    }
    #pragma unroll
    for (int off = 16; off; off >>= 1) sum += __shfl_xor(sum, off, 32);
    float mu = sum * (1.f / 512.f), s2 = 0.f;
    #pragma unroll
    for (int j = 0; j < 16; ++j) { float d = v[j] - mu; s2 += d * d; }
    #pragma unroll
    for (int off = 16; off; off >>= 1) s2 += __shfl_xor(s2, off, 32);
    float rstd = rsqrtf(s2 * (1.f / 512.f) + 1e-5f);
    #pragma unroll
    for (int j = 0; j < 16; ++j) {
        int c = j * 32 + lane;
        zb[(size_t)r * 512 + c] = f2bf((v[j] - mu) * rstd * gamma[c] + beta[c]);
    }
}

// ---------------------------------------------------------------------------
// Sequential GRU scan. grid.x = 2 (both directions concurrently, one WGP each).
// 1024 threads = 32 waves. Per step: gh = h(64x256,bf16 in swizzled LDS) @
// Whh^T(768x256) via 48 WMMAs/wave; f32 h lives in registers matching the
// WMMA C layout, so the GRU update is lane-local. Left direction scatter-
// writes through the masked-flip index.
// ---------------------------------------------------------------------------
__global__ __launch_bounds__(1024) void scan_kernel(
    const float* __restrict__ gi_r, const float* __restrict__ gi_l,
    const unsigned short* __restrict__ whh_r, const unsigned short* __restrict__ whh_l,
    const float* __restrict__ bhh_r, const float* __restrict__ bhh_l,
    const float* __restrict__ mask, const int* __restrict__ len,
    float* __restrict__ out)
{
    const int dir = blockIdx.x;
    const float* gi = dir ? gi_l : gi_r;
    const unsigned short* whh = dir ? whh_l : whh_r;
    const float* bhh = dir ? bhh_l : bhh_r;
    const int colbase = dir ? H_ : 0;

    __shared__ unsigned short hbuf[64][256];   // bf16 h, XOR-swizzled 16B chunks

    const int tid = threadIdx.x, lane = tid & 31, wave = tid >> 5;
    const int hi = lane >> 4, ln = lane & 15;

    for (int i = tid; i < 64 * 256; i += 1024) ((unsigned short*)hbuf)[i] = 0;

    // wave -> two position tiles p = 2w, 2w+1 ; mt = p&3 (row tile), jt = p>>2
    const int mt0 = (2 * wave) & 3, mt1 = (2 * wave + 1) & 3, jt = (2 * wave) >> 2;
    const int hc = jt * 16 + ln;

    float hreg[2][8];
    for (int pt = 0; pt < 2; ++pt)
        for (int v = 0; v < 8; ++v) hreg[pt][v] = 0.f;
    __syncthreads();

    for (int t = 0; t < S_; ++t) {
        v8f acc[2][3];
        for (int pt = 0; pt < 2; ++pt)
            for (int q = 0; q < 3; ++q)
                for (int e = 0; e < 8; ++e) acc[pt][q][e] = 0.f;

        #pragma unroll
        for (int k0 = 0; k0 < 256; k0 += 32) {
            // A frags from swizzled LDS (ds_load_b128, conflict-free)
            FragU fa0, fa1;
            {
                int r0 = mt0 * 16 + ln, r1 = mt1 * 16 + ln;
                int c0 = (k0 >> 3) + 2 * hi, c1 = c0 + 1;
                int s0 = r0 & 31, s1 = r1 & 31;
                fa0.q[0] = *(const u32x4*)&hbuf[r0][(c0 ^ s0) * 8];
                fa0.q[1] = *(const u32x4*)&hbuf[r0][(c1 ^ s0) * 8];
                fa1.q[0] = *(const u32x4*)&hbuf[r1][(c0 ^ s1) * 8];
                fa1.q[1] = *(const u32x4*)&hbuf[r1][(c1 ^ s1) * 8];
            }
            #pragma unroll
            for (int q = 0; q < 3; ++q) {
                v16bf b = load_b_frag(whh + ((size_t)(q * H_ + jt * 16)) * H_ + k0, H_);
                acc[0][q] = wmma_bf16(fa0.v, b, acc[0][q]);
                acc[1][q] = wmma_bf16(fa1.v, b, acc[1][q]);
            }
        }

        const float bh0 = bhh[hc], bh1 = bhh[H_ + hc], bh2 = bhh[2 * H_ + hc];
        #pragma unroll
        for (int pt = 0; pt < 2; ++pt) {
            const int mt = pt ? mt1 : mt0;
            #pragma unroll
            for (int v = 0; v < 8; ++v) {
                const int b = mt * 16 + v + 8 * hi;
                const size_t gib = ((size_t)t * B_ + b) * (3 * H_);
                float g0 = gi[gib + hc], g1 = gi[gib + H_ + hc], g2 = gi[gib + 2 * H_ + hc];
                float r = sigm(g0 + acc[pt][0][v] + bh0);
                float u = sigm(g1 + acc[pt][1][v] + bh1);
                float n = tanh_f(g2 + r * (acc[pt][2][v] + bh2));
                float hn = (1.f - u) * n + u * hreg[pt][v];
                hreg[pt][v] = hn;
                int lb = len[b];
                int s_out = dir ? ((t < lb) ? (lb - 1 - t) : t) : t;
                out[((size_t)s_out * B_ + b) * (2 * H_) + colbase + hc] =
                    mask[(size_t)b * S_ + s_out] * hn;
            }
        }
        __syncthreads();
        #pragma unroll
        for (int pt = 0; pt < 2; ++pt) {
            const int mt = pt ? mt1 : mt0;
            #pragma unroll
            for (int v = 0; v < 8; ++v) {
                const int b = mt * 16 + v + 8 * hi;
                const int phys = (((hc >> 3) ^ (b & 31)) * 8) + (hc & 7);
                hbuf[b][phys] = f2bf(hreg[pt][v]);
            }
        }
        __syncthreads();
    }
}

// ---------------------------------------------------------------------------
// Host orchestration
// ---------------------------------------------------------------------------
extern "C" void kernel_launch(void* const* d_in, const int* in_sizes, int n_in,
                              void* d_out, int out_size, void* d_ws, size_t ws_size,
                              hipStream_t stream)
{
    (void)in_sizes; (void)n_in; (void)out_size; (void)ws_size;
    const float* qt   = (const float*)d_in[0];
    const float* mask = (const float*)d_in[1];
    // params dict order: Wp,bp,Wpp,bpp,Wv,bv,Wg,bg,gamma,beta,Wih,bih,Whh,bhh
    auto gp = [&](int d, int k) -> const float* { return (const float*)d_in[2 + d * 14 + k]; };

    char* p = (char*)d_ws;
    auto alloc = [&](size_t bytes) -> void* {
        void* r = (void*)p; p += (bytes + 255) & ~(size_t)255; return r;
    };

    int* len            = (int*)alloc(B_ * sizeof(int));
    unsigned short* qtb = (unsigned short*)alloc((size_t)SB_ * I_ * 2);
    unsigned short* qtT = (unsigned short*)alloc((size_t)B_ * I_ * S_ * 2);
    const size_t wsz[5] = { (size_t)H_ * I_, (size_t)H_ * I_, 512ull * 512, 768ull * 512, 768ull * 256 };
    unsigned short* wbf[2][5];
    for (int d = 0; d < 2; ++d)
        for (int w = 0; w < 5; ++w) wbf[d][w] = (unsigned short*)alloc(wsz[w] * 2);
    float* wpB    = (float*)alloc((size_t)SB_ * H_ * 4);
    float* wppB   = (float*)alloc((size_t)SB_ * H_ * 4);
    float* scoref = (float*)alloc((size_t)SB_ * S_ * 4);
    unsigned short* alpha = (unsigned short*)alloc((size_t)SB_ * S_ * 2);
    float* z      = (float*)alloc((size_t)SB_ * 512 * 4);
    unsigned short* zb = (unsigned short*)alloc((size_t)SB_ * 512 * 2);
    float* gbuf   = (float*)alloc((size_t)SB_ * 512 * 4);
    float* giD[2] = { (float*)alloc((size_t)SB_ * 768 * 4), (float*)alloc((size_t)SB_ * 768 * 4) };

    lengths_kernel<<<1, 64, 0, stream>>>(mask, len);

    const int widx[5] = {0, 2, 6, 10, 12};   // Wp, Wpp, Wg, Wih, Whh
    for (int d = 0; d < 2; ++d) {
        for (int w = 0; w < 5; ++w) {
            int n = (int)wsz[w];
            cvt_bf16_kernel<<<(n + 255) / 256, 256, 0, stream>>>(gp(d, widx[w]), wbf[d][w], n);
        }
        cvt_qt_kernel<<<SB_, 256, 0, stream>>>(qt, len, d, qtb, qtT, z);
        // wp = qt@Wp^T + bp ; wpp = qt@Wpp^T + bpp
        gemm_bf16_nt<<<dim3(H_ / 128, SB_ / 128, 1), 256, 0, stream>>>(
            qtb, wbf[d][0], wpB, gp(d, 1), I_, I_, I_, H_, 0, 0, 0);
        gemm_bf16_nt<<<dim3(H_ / 128, SB_ / 128, 1), 256, 0, stream>>>(
            qtb, wbf[d][1], wppB, gp(d, 3), I_, I_, I_, H_, 0, 0, 0);
        score_kernel<<<dim3(B_, S_), 256, 0, stream>>>(wpB, wppB, gp(d, 4), gp(d, 5), scoref);
        softmax_kernel<<<SB_ / 8, 256, 0, stream>>>(scoref, mask, alpha);
        // h_alpha[t,b,:] : batched over b — alpha_b(384x384) @ qt_b^T, into z[...,I:2I]
        gemm_bf16_nt<<<dim3(I_ / 128, S_ / 128, B_), 256, 0, stream>>>(
            alpha, qtT, z + I_, nullptr, S_, SB_, S_, B_ * 2 * I_,
            (long long)S_, (long long)I_ * S_, (long long)(2 * I_));
        cvt_bf16_kernel<<<(SB_ * 512 + 255) / 256, 256, 0, stream>>>(z, zb, SB_ * 512);
        gemm_bf16_nt<<<dim3(512 / 128, SB_ / 128, 1), 256, 0, stream>>>(
            zb, wbf[d][2], gbuf, gp(d, 7), 512, 512, 512, 512, 0, 0, 0);
        gate_ln_kernel<<<SB_ / 8, 256, 0, stream>>>(z, gbuf, gp(d, 8), gp(d, 9), zb);
        gemm_bf16_nt<<<dim3(768 / 128, SB_ / 128, 1), 256, 0, stream>>>(
            zb, wbf[d][3], giD[d], gp(d, 11), 512, 512, 512, 768, 0, 0, 0);
    }
    scan_kernel<<<2, 1024, 0, stream>>>(giD[0], giD[1], wbf[0][4], wbf[1][4],
                                        gp(0, 13), gp(1, 13), mask, len, (float*)d_out);
}
